// RelationAwareMultiHeadedAttention_26963804685086
// MI455X (gfx1250) — matrix operand order repaired and checked by reading, not compiled, not run on hardware
//
#include <hip/hip_runtime.h>
#include <math.h>

// ---------------------------------------------------------------------------
// Problem constants (B,S,HID,H,NREL) = (4,512,1024,16,64), D=64
// ---------------------------------------------------------------------------
constexpr int cB    = 4;
constexpr int cS    = 512;
constexpr int cHID  = 1024;
constexpr int cH    = 16;
constexpr int cD    = 64;
constexpr int cNREL = 64;
constexpr int cNTOK = cB * cS;          // 2048 tokens

typedef _Float16 v16h __attribute__((ext_vector_type(16)));
typedef _Float16 v8h  __attribute__((ext_vector_type(8)));
typedef float    v8f  __attribute__((ext_vector_type(8)));

// ---------------------------------------------------------------------------
// WMMA helpers — fragment layouts with contiguous k = 16*(lane>>4) + i (wave32)
// ---------------------------------------------------------------------------
__device__ __forceinline__ v8f wmma_f16(v16h a, v16h b, v8f c) {
  // (neg_a, A, neg_b, B, c_mod, C, reuse_a, reuse_b)
  return __builtin_amdgcn_wmma_f32_16x16x32_f16(false, a, false, b, (short)0, c,
                                                false, false);
}

// A-matrix 16x32 f16 (MxK).  src points at tile origin (row0,k0), row-major,
// leading dim ld (elements).  Lane m=lane&15 holds row m; lanes<16 hold K
// 0..15, lanes>=16 hold K 16..31 (contiguous, matches B loaders).
__device__ __forceinline__ v16h load_a_f16(const _Float16* src, int ld, int lane) {
  const int m = lane & 15, g = lane >> 4;
  const _Float16* r = src + (size_t)m * ld + 16 * g;
  v8h lo = *(const v8h*)(r);
  v8h hi = *(const v8h*)(r + 8);
  v16h a;
#pragma unroll
  for (int i = 0; i < 8; ++i) { a[i] = lo[i]; a[8 + i] = hi[i]; }
  return a;
}

// Same A layout but gathering from an f32 source (LDS) with convert + scale.
__device__ __forceinline__ v16h load_a_f32cvt(const float* src, int ld, int lane,
                                              float scale) {
  const int m = lane & 15, g = lane >> 4;
  const float* r = src + (size_t)m * ld;
  v16h a;
#pragma unroll
  for (int i = 0; i < 8; ++i) {
    a[i]     = (_Float16)(r[16 * g + i] * scale);
    a[8 + i] = (_Float16)(r[16 * g + 8 + i] * scale);
  }
  return a;
}

// B-matrix 32x16 f16 (KxN) where B(k,n) = src[n*ld + k]  (N-major source:
// one memory row per output column — W rows / K rows / rel_k rows).
// One contiguous 32B load per lane.
__device__ __forceinline__ v16h load_b_nmajor(const _Float16* src, int ld, int lane) {
  const int n = lane & 15, g = lane >> 4;
  return *(const v16h*)(src + (size_t)n * ld + 16 * g);
}

// B-matrix 32x16 f16 (KxN) where B(k,n) = src[k*ld + n] (row-major K source —
// V tiles / rel_v).  Strided gather from LDS.
__device__ __forceinline__ v16h load_b_kmajor(const _Float16* src, int ld, int lane) {
  const int n = lane & 15, g = lane >> 4;
  v16h b;
#pragma unroll
  for (int i = 0; i < 16; ++i) b[i] = src[(size_t)(16 * g + i) * ld + n];
  return b;
}

// ---------------------------------------------------------------------------
// CDNA5 async global->LDS copy (16B per lane), tracked by ASYNCcnt.
// lds_byte_off = byte offset within the workgroup's dynamic LDS allocation
// (VDST per ISA holds the LDS offset; dynamic LDS starts at 0 here).
// ---------------------------------------------------------------------------
__device__ __forceinline__ void async_copy_b128(unsigned lds_byte_off,
                                                const void* gsrc) {
  asm volatile("global_load_async_to_lds_b128 %0, %1, off"
               :: "v"(lds_byte_off), "v"(gsrc)
               : "memory");
}
__device__ __forceinline__ void wait_asynccnt0() {
  asm volatile("s_wait_asynccnt 0x0" ::: "memory");
}

// ---------------------------------------------------------------------------
// f32 -> f16 convert (grid-stride)
// ---------------------------------------------------------------------------
__global__ void cvt_f32_to_f16(const float* __restrict__ src,
                               _Float16* __restrict__ dst, int n) {
  int i = blockIdx.x * blockDim.x + threadIdx.x;
  int stride = gridDim.x * blockDim.x;
  for (; i < n; i += stride) dst[i] = (_Float16)src[i];
}

// ---------------------------------------------------------------------------
// GEMM:  Out = X(MxK,f16) @ W(NxK,f16)^T + bias(N)      [X @ W.T + b]
// mode 0: OutF row-major (M,N) f32
// mode 1: OutH scattered to (B,H,S,D) f16  (row = b*S+s, col = h*D+d)
// Block = 256 thr = 8 waves; each wave: 16(M) x 64(N) strip, A reused x4.
// ---------------------------------------------------------------------------
__global__ void __launch_bounds__(256)
gemm_xwT_bias(const _Float16* __restrict__ X, const _Float16* __restrict__ W,
              const float* __restrict__ bias, float* __restrict__ OutF,
              _Float16* __restrict__ OutH, int M, int N, int K, int mode) {
  const int lane = threadIdx.x & 31;
  const int wave = threadIdx.x >> 5;
  const int m0 = (blockIdx.x * 8 + wave) * 16;
  const int n0 = blockIdx.y * 64;
  if (m0 >= M) return;

  const v8f vzero = {0.f, 0.f, 0.f, 0.f, 0.f, 0.f, 0.f, 0.f};
  v8f acc[4] = {vzero, vzero, vzero, vzero};

  for (int k0 = 0; k0 < K; k0 += 32) {
    v16h a = load_a_f16(X + (size_t)m0 * K + k0, K, lane);
#pragma unroll
    for (int nt = 0; nt < 4; ++nt) {
      v16h b = load_b_nmajor(W + (size_t)(n0 + nt * 16) * K + k0, K, lane);
      acc[nt] = wmma_f16(a, b, acc[nt]);
    }
  }

  const int g = lane >> 4, n = lane & 15;
#pragma unroll
  for (int nt = 0; nt < 4; ++nt) {
#pragma unroll
    for (int r = 0; r < 8; ++r) {
      const int row = m0 + r + 8 * g;
      const int col = n0 + nt * 16 + n;
      const float val = acc[nt][r] + bias[col];
      if (mode == 0) {
        OutF[(size_t)row * N + col] = val;
      } else {
        const int b_ = row >> 9, s_ = row & (cS - 1);     // S = 512
        const int h_ = col >> 6, d_ = col & (cD - 1);     // D = 64
        OutH[(((size_t)b_ * cH + h_) * cS + s_) * cD + d_] = (_Float16)val;
      }
    }
  }
}

// ---------------------------------------------------------------------------
// Fused relation-aware attention.  All operands are f16 (projections stored
// f16 by the GEMM above), so K/V/rel tables stream to LDS with CDNA5 async
// global->LDS copies (ASYNCcnt), no VGPR staging.
//   grid.x = B*H*(S/64); block = 128 thr (4 waves); wave w owns 16 queries.
// LDS (dynamic, 304KB): Kh[S*D]f16 | Vh[S*D]f16 | rkh[64*64]f16 | rvh f16 |
//   probs[4][16][S]f32 | qrel[4][16][64]f32 | wbuf[4][16][64]f32
// scores = (Q·K^T + 0.25*gather(Q·relK^T, ids)) / 8 ; bias -1e4 where id==0
// out    = P@V + (0.25*hist(P,ids)) @ relV        (all WMMA)
// ---------------------------------------------------------------------------
__global__ void __launch_bounds__(128)
relattn_kernel(const _Float16* __restrict__ qp, const _Float16* __restrict__ kp,
               const _Float16* __restrict__ vp, const int* __restrict__ mask,
               const _Float16* __restrict__ relk, const _Float16* __restrict__ relv,
               _Float16* __restrict__ attn) {
  extern __shared__ char smem[];
  _Float16* Kh  = (_Float16*)smem;                    // 32768 halfs
  _Float16* Vh  = Kh + cS * cD;                       // 32768 halfs
  _Float16* rkh = Vh + cS * cD;                       // 4096 halfs
  _Float16* rvh = rkh + cNREL * cD;                   // 4096 halfs
  float* probs  = (float*)(rvh + cNREL * cD);         // 4*16*512 f32
  float* qrel   = probs + 4 * 16 * cS;                // 4*16*64 f32
  float* wbuf   = qrel + 4 * 16 * cD;                 // 4*16*64 f32

  const int tid = threadIdx.x;
  const int lane = tid & 31;
  const int wave = tid >> 5;
  const int qblkPerBH = cS / 64;                       // 8
  const int bh   = blockIdx.x / qblkPerBH;
  const int qblk = blockIdx.x % qblkPerBH;
  const int b = bh / cH, h = bh % cH;

  // ---- async stage: K,V (64KB each) + rel tables (8KB each) into LDS ----
  {
    const _Float16* gk = kp + (size_t)bh * cS * cD;
    const _Float16* gv = vp + (size_t)bh * cS * cD;
    const unsigned offK  = (unsigned)((char*)Kh  - smem);
    const unsigned offV  = (unsigned)((char*)Vh  - smem);
    const unsigned offRK = (unsigned)((char*)rkh - smem);
    const unsigned offRV = (unsigned)((char*)rvh - smem);
    for (int i = tid * 8; i < cS * cD; i += 128 * 8) {   // 16B chunks
      async_copy_b128(offK + i * 2, gk + i);
      async_copy_b128(offV + i * 2, gv + i);
    }
    for (int i = tid * 8; i < cNREL * cD; i += 128 * 8) {
      async_copy_b128(offRK + i * 2, relk + i);
      async_copy_b128(offRV + i * 2, relv + i);
    }
    wait_asynccnt0();
  }
  __syncthreads();

  const int q0 = qblk * 64 + wave * 16;                // first query of my tile
  const _Float16* qbase = qp + ((size_t)bh * cS + q0) * cD;
  float* qrelW  = qrel  + wave * 16 * cD;
  float* probsW = probs + wave * 16 * cS;
  float* wbufW  = wbuf  + wave * 16 * cD;
  const int* mrow = mask + ((size_t)b * cS + q0) * cS; // mask[b,0,q0+.. ,:]

  // Q tile as two f16 A-fragments (K-dim = D = 64), contiguous global loads
  v16h aq0 = load_a_f16(qbase + 0,  cD, lane);
  v16h aq1 = load_a_f16(qbase + 32, cD, lane);

  const v8f vzero = {0.f, 0.f, 0.f, 0.f, 0.f, 0.f, 0.f, 0.f};
  const int g = lane >> 4, nlo = lane & 15;

  // ---- qrel = Q_tile(16x64) @ rel_k_table^T(64x64) ----
#pragma unroll
  for (int nt = 0; nt < 4; ++nt) {
    v8f acc = vzero;
    acc = wmma_f16(aq0, load_b_nmajor(rkh + (size_t)(nt * 16) * cD + 0,  cD, lane), acc);
    acc = wmma_f16(aq1, load_b_nmajor(rkh + (size_t)(nt * 16) * cD + 32, cD, lane), acc);
#pragma unroll
    for (int r = 0; r < 8; ++r)
      qrelW[(r + 8 * g) * cD + nt * 16 + nlo] = acc[r];
  }

  // ---- scores: 32 key-tiles of 16, each two WMMAs over D ----
  for (int kt = 0; kt < cS / 16; ++kt) {
    v8f acc = vzero;
    acc = wmma_f16(aq0, load_b_nmajor(Kh + (size_t)(kt * 16) * cD + 0,  cD, lane), acc);
    acc = wmma_f16(aq1, load_b_nmajor(Kh + (size_t)(kt * 16) * cD + 32, cD, lane), acc);
    const int key = kt * 16 + nlo;
#pragma unroll
    for (int r = 0; r < 8; ++r) {
      const int q = r + 8 * g;
      const int id = mrow[(size_t)q * cS + key];
      float s = (acc[r] + 0.25f * qrelW[q * cD + id]) * 0.125f;  // /sqrt(64)
      if (id == 0) s -= 10000.0f;
      probsW[(size_t)q * cS + key] = s;
    }
  }

  // ---- zero relation-histogram buffer ----
  for (int i = lane; i < 16 * cD; i += 32) wbufW[i] = 0.0f;

  // ---- softmax per query row (32 lanes cooperate per row) + histogram ----
  for (int q = 0; q < 16; ++q) {
    float* row = probsW + (size_t)q * cS;
    float mx = -3.4e38f;
#pragma unroll
    for (int i = 0; i < cS / 32; ++i) mx = fmaxf(mx, row[lane + 32 * i]);
    for (int off = 16; off > 0; off >>= 1) mx = fmaxf(mx, __shfl_xor(mx, off, 32));
    float sum = 0.0f;
#pragma unroll
    for (int i = 0; i < cS / 32; ++i) {
      const float e = __expf(row[lane + 32 * i] - mx);
      row[lane + 32 * i] = e;
      sum += e;
    }
    for (int off = 16; off > 0; off >>= 1) sum += __shfl_xor(sum, off, 32);
    const float inv = 1.0f / sum;
#pragma unroll
    for (int i = 0; i < cS / 32; ++i) {
      const int key = lane + 32 * i;
      const float p = row[key] * inv;
      row[key] = p;
      atomicAdd(&wbufW[q * cD + mrow[(size_t)q * cS + key]], p);  // ds_add_f32
    }
  }

  // ---- out = P(16x512)@V(512x64) + (0.25*w)(16x64)@relV(64x64) ----
  v8f acc[4] = {vzero, vzero, vzero, vzero};
  for (int ks = 0; ks < cS / 32; ++ks) {
    v16h ap = load_a_f32cvt(probsW + ks * 32, cS, lane, 1.0f);
#pragma unroll
    for (int nt = 0; nt < 4; ++nt)
      acc[nt] = wmma_f16(ap, load_b_kmajor(Vh + (size_t)(ks * 32) * cD + nt * 16, cD, lane),
                         acc[nt]);
  }
  {
    v16h aw0 = load_a_f32cvt(wbufW + 0,  cD, lane, 0.25f);
    v16h aw1 = load_a_f32cvt(wbufW + 32, cD, lane, 0.25f);
#pragma unroll
    for (int nt = 0; nt < 4; ++nt) {
      acc[nt] = wmma_f16(aw0, load_b_kmajor(rvh + 0 * cD + nt * 16,  cD, lane), acc[nt]);
      acc[nt] = wmma_f16(aw1, load_b_kmajor(rvh + 32 * cD + nt * 16, cD, lane), acc[nt]);
    }
  }

  // ---- write attn (B,S,HID) f16 (feeds f16 output-projection GEMM) ----
#pragma unroll
  for (int nt = 0; nt < 4; ++nt) {
#pragma unroll
    for (int r = 0; r < 8; ++r) {
      const int q = q0 + r + 8 * g;
      attn[((size_t)b * cS + q) * cHID + h * cD + nt * 16 + nlo] =
          (_Float16)acc[nt][r];
    }
  }
}

// ---------------------------------------------------------------------------
// Host-side orchestration
// ---------------------------------------------------------------------------
extern "C" void kernel_launch(void* const* d_in, const int* in_sizes, int n_in,
                              void* d_out, int out_size, void* d_ws, size_t ws_size,
                              hipStream_t stream) {
  const float* q_in = (const float*)d_in[0];
  const float* k_in = (const float*)d_in[1];
  const float* v_in = (const float*)d_in[2];
  const int*   mask = (const int*)d_in[3];
  const float* Wq = (const float*)d_in[4];
  const float* bq = (const float*)d_in[5];
  const float* Wk = (const float*)d_in[6];
  const float* bk = (const float*)d_in[7];
  const float* Wv = (const float*)d_in[8];
  const float* bv = (const float*)d_in[9];
  const float* Wo = (const float*)d_in[10];
  const float* bo = (const float*)d_in[11];
  const float* rel_k = (const float*)d_in[12];
  const float* rel_v = (const float*)d_in[13];
  float* out = (float*)d_out;

  // workspace carve (256B aligned)
  char* ws = (char*)d_ws;
  size_t off = 0;
  auto carve = [&](size_t bytes) -> void* {
    void* p = ws + off;
    off += (bytes + 255) & ~(size_t)255;
    return p;
  };
  const size_t nTok = (size_t)cNTOK * cHID;   // 2M elems
  const size_t nWt  = (size_t)cHID * cHID;    // 1M elems
  _Float16* qh  = (_Float16*)carve(nTok * 2);
  _Float16* kh  = (_Float16*)carve(nTok * 2);
  _Float16* vh  = (_Float16*)carve(nTok * 2);
  _Float16* wqh = (_Float16*)carve(nWt * 2);
  _Float16* wkh = (_Float16*)carve(nWt * 2);
  _Float16* wvh = (_Float16*)carve(nWt * 2);
  _Float16* woh = (_Float16*)carve(nWt * 2);
  _Float16* rkh = (_Float16*)carve((size_t)cNREL * cD * 2);
  _Float16* rvh = (_Float16*)carve((size_t)cNREL * cD * 2);
  _Float16* qp    = (_Float16*)carve(nTok * 2);  // (B,H,S,D) f16
  _Float16* kp    = (_Float16*)carve(nTok * 2);
  _Float16* vp    = (_Float16*)carve(nTok * 2);
  _Float16* attnh = (_Float16*)carve(nTok * 2);  // (B,S,HID) f16

  // 1) f32 -> f16 converts (inputs + weights + rel tables)
  cvt_f32_to_f16<<<2048, 256, 0, stream>>>(q_in, qh, (int)nTok);
  cvt_f32_to_f16<<<2048, 256, 0, stream>>>(k_in, kh, (int)nTok);
  cvt_f32_to_f16<<<2048, 256, 0, stream>>>(v_in, vh, (int)nTok);
  cvt_f32_to_f16<<<1024, 256, 0, stream>>>(Wq, wqh, (int)nWt);
  cvt_f32_to_f16<<<1024, 256, 0, stream>>>(Wk, wkh, (int)nWt);
  cvt_f32_to_f16<<<1024, 256, 0, stream>>>(Wv, wvh, (int)nWt);
  cvt_f32_to_f16<<<1024, 256, 0, stream>>>(Wo, woh, (int)nWt);
  cvt_f32_to_f16<<<16, 256, 0, stream>>>(rel_k, rkh, cNREL * cD);
  cvt_f32_to_f16<<<16, 256, 0, stream>>>(rel_v, rvh, cNREL * cD);

  // 2) Q/K/V projections: (2048x1024) @ W^T -> (B,H,S,D) f16
  dim3 ggrid(cNTOK / 128, cHID / 64);
  gemm_xwT_bias<<<ggrid, 256, 0, stream>>>(qh, wqh, bq, nullptr, qp,
                                           cNTOK, cHID, cHID, 1);
  gemm_xwT_bias<<<ggrid, 256, 0, stream>>>(kh, wkh, bk, nullptr, kp,
                                           cNTOK, cHID, cHID, 1);
  gemm_xwT_bias<<<ggrid, 256, 0, stream>>>(vh, wvh, bv, nullptr, vp,
                                           cNTOK, cHID, cHID, 1);

  // 3) fused relation-aware attention (async LDS staging + WMMA)
  const size_t ldsBytes = (size_t)(cS * cD) * 2 * 2      // Kh + Vh
                        + (size_t)(cNREL * cD) * 2 * 2   // rkh + rvh
                        + (size_t)4 * 16 * cS * 4        // probs
                        + (size_t)4 * 16 * cD * 4        // qrel
                        + (size_t)4 * 16 * cD * 4;       // wbuf   = 311296 B
  relattn_kernel<<<cB * cH * (cS / 64), 128, ldsBytes, stream>>>(
      qp, kp, vp, mask, rkh, rvh, attnh);

  // 4) output projection: attn @ Wo^T + bo -> d_out (B,S,HID) f32
  gemm_xwT_bias<<<ggrid, 256, 0, stream>>>(attnh, woh, bo, out, nullptr,
                                           cNTOK, cHID, cHID, 0);
}
